// Block_11785390260704
// MI455X (gfx1250) — matrix operand order repaired and checked by reading, not compiled, not run on hardware
//
#include <hip/hip_runtime.h>
#include <hip/hip_bf16.h>
#include <math.h>

// ---------------- problem constants ----------------
#define BB 2
#define SS 2048
#define HH 1024
#define NHD 16
#define NOPE 32
#define ROPE 32
#define HD 64
#define QR 128
#define KVR 128
#define EE 8
#define KK 2
#define II 384
#define ESH 2
#define TT (BB * SS)   // 4096

typedef __attribute__((ext_vector_type(16))) __bf16 v16bf;
typedef __attribute__((ext_vector_type(8)))  float  v8f;

union U16 { v16bf v; unsigned int u[8]; };

__device__ __forceinline__ __bf16 f2bf(float f) {
  unsigned int u = __builtin_bit_cast(unsigned int, f);
  u = (u + 0x7FFFu + ((u >> 16) & 1u)) >> 16;
  unsigned short h = (unsigned short)u;
  return __builtin_bit_cast(__bf16, h);
}

__device__ __forceinline__ float frcp(float x) { return __builtin_amdgcn_rcpf(x); }

__device__ __forceinline__ v8f wmma_bf16(const U16& a, const U16& b, v8f c) {
  return __builtin_amdgcn_wmma_f32_16x16x32_bf16(false, a.v, false, b.v,
                                                 (short)0, c, false, false);
}

// ---------------- f32 -> bf16 convert ----------------
__global__ __launch_bounds__(256) void k_f32_to_bf16(const float* __restrict__ s,
                                                     __bf16* __restrict__ d, size_t n) {
  for (size_t i = (size_t)blockIdx.x * 256 + threadIdx.x; i < n; i += (size_t)gridDim.x * 256)
    d[i] = f2bf(s[i]);
}

// ---------------- RMSNorm (rows of length Hdim) ----------------
__global__ __launch_bounds__(256) void k_rms(const float* __restrict__ x,
                                             const float* __restrict__ w,
                                             __bf16* __restrict__ out_bf,
                                             float* __restrict__ out_f32,
                                             int Hdim) {
  __shared__ float red[256];
  int t = blockIdx.x;
  const float* row = x + (size_t)t * Hdim;
  float s = 0.f;
  for (int i = threadIdx.x; i < Hdim; i += 256) { float v = row[i]; s += v * v; }
  red[threadIdx.x] = s;
  __syncthreads();
  for (int st = 128; st > 0; st >>= 1) {
    if ((int)threadIdx.x < st) red[threadIdx.x] += red[threadIdx.x + st];
    __syncthreads();
  }
  float inv = rsqrtf(red[0] / (float)Hdim + 1e-5f);
  for (int i = threadIdx.x; i < Hdim; i += 256) {
    float v = row[i] * inv * w[i];
    out_bf[(size_t)t * Hdim + i] = f2bf(v);
    if (out_f32) out_f32[(size_t)t * Hdim + i] = v;
  }
}

// ---------------- generic bf16 WMMA GEMM ----------------
// C[M,N] (+)= A[M,K]*B[K,N]; block tile 128x64, 8 waves (4x2), wave tile 32x32
// MODE 0: C = AB ; MODE 1: C = addsrc + AB ; MODE 2: C += rowscale[m]*AB
template <int MODE>
__global__ __launch_bounds__(256) void k_gemm_bf16(const __bf16* __restrict__ A, int lda,
                                                   const __bf16* __restrict__ B, int ldb,
                                                   float* __restrict__ C, int ldc,
                                                   int M, int N, int Kdim,
                                                   const float* __restrict__ addsrc, int ldadd,
                                                   const float* __restrict__ rowscale, int rstride) {
  __shared__ __align__(16) __bf16 ldsA[128 * 32];
  __shared__ __align__(16) __bf16 ldsB[64 * 32]; // [n][k]
  const int tid = threadIdx.x;
  const int lane = tid & 31, wid = tid >> 5;
  const int half = lane >> 4, ln = lane & 15;
  const int m0 = blockIdx.x * 128;
  const int n0 = blockIdx.y * 64;
  const int wm = (wid & 3) * 32;
  const int wn = (wid >> 2) * 32;
  v8f acc[2][2];
#pragma unroll
  for (int i = 0; i < 2; ++i)
#pragma unroll
    for (int j = 0; j < 2; ++j) acc[i][j] = (v8f){0.f,0.f,0.f,0.f,0.f,0.f,0.f,0.f};

  for (int k0 = 0; k0 < Kdim; k0 += 32) {
    int kp = (k0 + 32 < Kdim) ? (k0 + 32) : k0;
#pragma unroll
    for (int c2 = 0; c2 < 2; ++c2) { // A tile 128x32 (16 bf16/thread)
      int e = tid * 16 + c2 * 8;
      int r = e >> 5, c = e & 31;
      *(uint4*)(&ldsA[e]) = *(const uint4*)(A + (size_t)(m0 + r) * lda + k0 + c);
      __builtin_prefetch(A + (size_t)(m0 + r) * lda + kp + c, 0, 1);
    }
#pragma unroll
    for (int i = 0; i < 8; ++i) { // B tile 32x64 -> [n][k]
      int e = tid + i * 256;
      int nn = e & 63, kk = e >> 6;
      ldsB[nn * 32 + kk] = B[(size_t)(k0 + kk) * ldb + n0 + nn];
    }
    __builtin_prefetch(B + (size_t)(kp + (tid >> 6)) * ldb + n0 + (tid & 63), 0, 1);
    __syncthreads();

    U16 a0, a1, b0, b1;
    {
      int khi = half * 4;
      const unsigned int* r0 = (const unsigned int*)(&ldsA[(wm + ln) * 32]);
      const unsigned int* r1 = (const unsigned int*)(&ldsA[(wm + 16 + ln) * 32]);
#pragma unroll
      for (int p = 0; p < 8; ++p) {
        int idx = (p & 3) + ((p >> 2) * 8) + khi;
        a0.u[p] = r0[idx];
        a1.u[p] = r1[idx];
      }
      int kb = half * 4;
      const unsigned int* s0 = (const unsigned int*)(&ldsB[(wn + ln) * 32]);
      const unsigned int* s1 = (const unsigned int*)(&ldsB[(wn + 16 + ln) * 32]);
#pragma unroll
      for (int p = 0; p < 8; ++p) { int q = (p & 3) + ((p >> 2) * 8) + kb; b0.u[p] = s0[q]; b1.u[p] = s1[q]; }
    }
    acc[0][0] = wmma_bf16(a0, b0, acc[0][0]);
    acc[0][1] = wmma_bf16(a0, b1, acc[0][1]);
    acc[1][0] = wmma_bf16(a1, b0, acc[1][0]);
    acc[1][1] = wmma_bf16(a1, b1, acc[1][1]);
    __syncthreads();
  }

#pragma unroll
  for (int i = 0; i < 2; ++i)
#pragma unroll
    for (int j = 0; j < 2; ++j) {
      int mb = m0 + wm + i * 16 + half * 8;
      int nc = n0 + wn + j * 16 + ln;
#pragma unroll
      for (int r = 0; r < 8; ++r) {
        int mr = mb + r;
        float v = acc[i][j][r];
        size_t off = (size_t)mr * ldc + nc;
        if (MODE == 0) {
          C[off] = v;
        } else if (MODE == 1) {
          C[off] = v + addsrc[(size_t)mr * ldadd + nc];
        } else {
          float sc = rowscale ? rowscale[(size_t)mr * rstride] : 1.0f;
          C[off] += sc * v;
        }
      }
    }
}

// ---------------- fused dual GEMM + SiLU*up (expert gate/up) ----------------
// act[M,N] = bf16( silu(A*Bg) * (A*Bu) ) ; same tiling as k_gemm_bf16
__global__ __launch_bounds__(256) void k_gemm_dual_silu(const __bf16* __restrict__ A, int lda,
                                                        const __bf16* __restrict__ Bg,
                                                        const __bf16* __restrict__ Bu, int ldb,
                                                        __bf16* __restrict__ act, int ldo,
                                                        int M, int N, int Kdim) {
  __shared__ __align__(16) __bf16 ldsA[128 * 32];
  __shared__ __align__(16) __bf16 ldsBg[64 * 32];
  __shared__ __align__(16) __bf16 ldsBu[64 * 32];
  const int tid = threadIdx.x;
  const int lane = tid & 31, wid = tid >> 5;
  const int half = lane >> 4, ln = lane & 15;
  const int m0 = blockIdx.x * 128;
  const int n0 = blockIdx.y * 64;
  const int wm = (wid & 3) * 32;
  const int wn = (wid >> 2) * 32;
  v8f ag[2][2], au[2][2];
#pragma unroll
  for (int i = 0; i < 2; ++i)
#pragma unroll
    for (int j = 0; j < 2; ++j) {
      ag[i][j] = (v8f){0.f,0.f,0.f,0.f,0.f,0.f,0.f,0.f};
      au[i][j] = (v8f){0.f,0.f,0.f,0.f,0.f,0.f,0.f,0.f};
    }

  for (int k0 = 0; k0 < Kdim; k0 += 32) {
    int kp = (k0 + 32 < Kdim) ? (k0 + 32) : k0;
#pragma unroll
    for (int c2 = 0; c2 < 2; ++c2) {
      int e = tid * 16 + c2 * 8;
      int r = e >> 5, c = e & 31;
      *(uint4*)(&ldsA[e]) = *(const uint4*)(A + (size_t)(m0 + r) * lda + k0 + c);
      __builtin_prefetch(A + (size_t)(m0 + r) * lda + kp + c, 0, 1);
    }
#pragma unroll
    for (int i = 0; i < 8; ++i) {
      int e = tid + i * 256;
      int nn = e & 63, kk = e >> 6;
      size_t go = (size_t)(k0 + kk) * ldb + n0 + nn;
      ldsBg[nn * 32 + kk] = Bg[go];
      ldsBu[nn * 32 + kk] = Bu[go];
    }
    __builtin_prefetch(Bg + (size_t)(kp + (tid >> 6)) * ldb + n0 + (tid & 63), 0, 1);
    __builtin_prefetch(Bu + (size_t)(kp + (tid >> 6)) * ldb + n0 + (tid & 63), 0, 1);
    __syncthreads();

    U16 a0, a1, g0, g1, u0, u1;
    {
      int khi = half * 4;
      const unsigned int* r0 = (const unsigned int*)(&ldsA[(wm + ln) * 32]);
      const unsigned int* r1 = (const unsigned int*)(&ldsA[(wm + 16 + ln) * 32]);
#pragma unroll
      for (int p = 0; p < 8; ++p) {
        int idx = (p & 3) + ((p >> 2) * 8) + khi;
        a0.u[p] = r0[idx];
        a1.u[p] = r1[idx];
      }
      int kb = half * 4;
      const unsigned int* sg0 = (const unsigned int*)(&ldsBg[(wn + ln) * 32]);
      const unsigned int* sg1 = (const unsigned int*)(&ldsBg[(wn + 16 + ln) * 32]);
      const unsigned int* su0 = (const unsigned int*)(&ldsBu[(wn + ln) * 32]);
      const unsigned int* su1 = (const unsigned int*)(&ldsBu[(wn + 16 + ln) * 32]);
#pragma unroll
      for (int p = 0; p < 8; ++p) {
        int q = (p & 3) + ((p >> 2) * 8) + kb;
        g0.u[p] = sg0[q]; g1.u[p] = sg1[q];
        u0.u[p] = su0[q]; u1.u[p] = su1[q];
      }
    }
    ag[0][0] = wmma_bf16(a0, g0, ag[0][0]);
    ag[0][1] = wmma_bf16(a0, g1, ag[0][1]);
    ag[1][0] = wmma_bf16(a1, g0, ag[1][0]);
    ag[1][1] = wmma_bf16(a1, g1, ag[1][1]);
    au[0][0] = wmma_bf16(a0, u0, au[0][0]);
    au[0][1] = wmma_bf16(a0, u1, au[0][1]);
    au[1][0] = wmma_bf16(a1, u0, au[1][0]);
    au[1][1] = wmma_bf16(a1, u1, au[1][1]);
    __syncthreads();
  }

#pragma unroll
  for (int i = 0; i < 2; ++i)
#pragma unroll
    for (int j = 0; j < 2; ++j) {
      int mb = m0 + wm + i * 16 + half * 8;
      int nc = n0 + wn + j * 16 + ln;
#pragma unroll
      for (int r = 0; r < 8; ++r) {
        float gv = ag[i][j][r];
        float s = gv * frcp(1.0f + __expf(-gv));   // v_rcp_f32, not exact-div chain
        act[(size_t)(mb + r) * ldo + nc] = f2bf(s * au[i][j][r]);
      }
    }
}

// ---------------- assemble q/k heads with RoPE ----------------
__global__ __launch_bounds__(256) void k_assemble_qk(const float* __restrict__ nope,
                                                     const float* __restrict__ rope,
                                                     const float* __restrict__ cosb,
                                                     const float* __restrict__ sinb,
                                                     __bf16* __restrict__ out) {
  size_t i = (size_t)blockIdx.x * 256 + threadIdx.x;
  int d = (int)(i & 63);
  int s = (int)((i >> 6) & 2047);
  int h = (int)((i >> 17) & 15);
  int b = (int)(i >> 21);
  size_t t = (size_t)b * SS + s;
  float val;
  if (d < 32) {
    val = nope[t * 512 + h * 32 + d];
  } else {
    int dd = d - 32;
    float xv = rope[t * 512 + h * 32 + dd];
    float xr = (dd < 16) ? -rope[t * 512 + h * 32 + dd + 16]
                         :  rope[t * 512 + h * 32 + dd - 16];
    val = xv * cosb[s * 32 + dd] + xr * sinb[s * 32 + dd];
  }
  out[i] = f2bf(val);
}

__global__ __launch_bounds__(256) void k_assemble_v(const float* __restrict__ vf,
                                                    __bf16* __restrict__ out) {
  size_t i = (size_t)blockIdx.x * 256 + threadIdx.x;
  int d = (int)(i & 63);
  int s = (int)((i >> 6) & 2047);
  int h = (int)((i >> 17) & 15);
  int b = (int)(i >> 21);
  size_t t = (size_t)b * SS + s;
  out[i] = f2bf(vf[t * 1024 + h * 64 + d]);
}

// ---------------- flash attention (streaming softmax, WMMA) ----------------
__global__ __launch_bounds__(256) void k_mla_attn(const __bf16* __restrict__ Q,
                                                  const __bf16* __restrict__ Kt,
                                                  const __bf16* __restrict__ V,
                                                  const float* __restrict__ mask,
                                                  __bf16* __restrict__ O) {
  __shared__ __align__(16) __bf16 ldsK[32 * 64];      // [key][hd]
  __shared__ __align__(16) __bf16 ldsV[64 * 32];      // [hd][key]
  __shared__ float  ldsS[8][16 * 32];
  __shared__ __align__(4) __bf16 ldsP[8][16 * 32];
  __shared__ float  ldsR[8][16];

  const int tid = threadIdx.x;
  const int lane = tid & 31, wid = tid >> 5;
  const int half = lane >> 4, ln = lane & 15;
  const int qt = blockIdx.x, h = blockIdx.y, b = blockIdx.z;

  const size_t bh = ((size_t)b * NHD + h) * SS * HD;
  const __bf16* Qb = Q + bh;
  const __bf16* Kb = Kt + bh;
  const __bf16* Vb = V + bh;
  const int qbase = qt * 128 + wid * 16;

  U16 qf0, qf1;
  {
    const unsigned int* rowQ = (const unsigned int*)(Qb + (size_t)(qbase + ln) * HD);
    int khi = half * 4;
#pragma unroll
    for (int p = 0; p < 8; ++p) {
      int idx = (p & 3) + ((p >> 2) * 8) + khi;
      qf0.u[p] = rowQ[idx];
      qf1.u[p] = rowQ[16 + idx];
    }
  }

  float mrow = -1e30f, lrow = 0.f;
  v8f o0 = {0.f,0.f,0.f,0.f,0.f,0.f,0.f,0.f};
  v8f o1 = o0, o2 = o0, o3 = o0;

  for (int kc = 0; kc < SS; kc += 32) {
    int kcn = (kc + 32 < SS) ? kc + 32 : kc;
    {
      int e = tid * 8;
      int r = e >> 6, c = e & 63;
      *(uint4*)(&ldsK[e]) = *(const uint4*)(Kb + (size_t)(kc + r) * HD + c);
      __builtin_prefetch(Kb + (size_t)(kcn + r) * HD + c, 0, 1);
      __builtin_prefetch(Vb + (size_t)(kcn + r) * HD + c, 0, 1);
    }
#pragma unroll
    for (int i = 0; i < 8; ++i) {
      int e = tid + i * 256;
      int hd = e & 63, ky = e >> 6;
      ldsV[hd * 32 + ky] = Vb[(size_t)(kc + ky) * HD + hd];
    }
    __syncthreads();

#pragma unroll
    for (int t2 = 0; t2 < 2; ++t2) {
      U16 b0, b1;
      const unsigned int* rowK = (const unsigned int*)(&ldsK[(t2 * 16 + ln) * 64]);
      int kb = half * 4;
#pragma unroll
      for (int p = 0; p < 8; ++p) { int q = (p & 3) + ((p >> 2) * 8) + kb; b0.u[p] = rowK[q]; b1.u[p] = rowK[16 + q]; }
      v8f s = {0.f,0.f,0.f,0.f,0.f,0.f,0.f,0.f};
      s = wmma_bf16(qf0, b0, s);
      s = wmma_bf16(qf1, b1, s);
#pragma unroll
      for (int r = 0; r < 8; ++r) {
        int m = r + half * 8;
        int key = kc + t2 * 16 + ln;
        float val = s[r] * 0.125f + mask[(size_t)(qbase + m) * SS + key];
        ldsS[wid][m * 32 + t2 * 16 + ln] = val;
      }
    }
    __syncthreads();

    if (lane < 16) {
      float* srow = &ldsS[wid][lane * 32];
      float mx = mrow;
#pragma unroll
      for (int j = 0; j < 32; ++j) mx = fmaxf(mx, srow[j]);
      float corr = __expf(mrow - mx);
      float sum = 0.f;
#pragma unroll
      for (int j = 0; j < 32; ++j) {
        float p = __expf(srow[j] - mx);
        sum += p;
        ldsP[wid][lane * 32 + j] = f2bf(p);
      }
      lrow = lrow * corr + sum;
      mrow = mx;
      ldsR[wid][lane] = corr;
    }
    __syncthreads();

#pragma unroll
    for (int r = 0; r < 8; ++r) {
      float c = ldsR[wid][r + half * 8];
      o0[r] *= c; o1[r] *= c; o2[r] *= c; o3[r] *= c;
    }
    U16 pf;
    {
      const unsigned int* rowP = (const unsigned int*)(&ldsP[wid][ln * 32]);
      int khi = half * 4;
#pragma unroll
      for (int p = 0; p < 8; ++p) pf.u[p] = rowP[(p & 3) + ((p >> 2) * 8) + khi];
    }
#pragma unroll
    for (int nt = 0; nt < 4; ++nt) {
      U16 vf;
      const unsigned int* rowV = (const unsigned int*)(&ldsV[(nt * 16 + ln) * 32]);
#pragma unroll
      for (int p = 0; p < 8; ++p) vf.u[p] = rowV[(p & 3) + ((p >> 2) * 8) + half * 4];
      v8f* oc = (nt == 0) ? &o0 : (nt == 1) ? &o1 : (nt == 2) ? &o2 : &o3;
      *oc = wmma_bf16(pf, vf, *oc);
    }
    __syncthreads();
  }

  if (lane < 16) ldsR[wid][lane] = lrow;
  __syncthreads();
#pragma unroll
  for (int r = 0; r < 8; ++r) {
    int m = r + half * 8;
    float inv = frcp(ldsR[wid][m]);               // v_rcp_f32
    int q = qbase + m;
    size_t rowoff = ((size_t)b * SS + q) * (NHD * HD) + (size_t)h * HD;
    O[rowoff +  0 + ln] = f2bf(o0[r] * inv);
    O[rowoff + 16 + ln] = f2bf(o1[r] * inv);
    O[rowoff + 32 + ln] = f2bf(o2[r] * inv);
    O[rowoff + 48 + ln] = f2bf(o3[r] * inv);
  }
}

// ---------------- MoE router / top-k ----------------
__global__ __launch_bounds__(256) void k_router(const float* __restrict__ hn,
                                                const float* __restrict__ w,
                                                const float* __restrict__ bias,
                                                float* __restrict__ probs) {
  int idx = blockIdx.x * 256 + threadIdx.x; // T*E
  int e = idx & 7, t = idx >> 3;
  float s = bias[e];
  for (int k = 0; k < HH; ++k) s += hn[(size_t)t * HH + k] * w[k * EE + e];
  probs[idx] = frcp(1.0f + __expf(-s));
}

__global__ void k_zero8(float* p) { if (threadIdx.x < EE) p[threadIdx.x] = 0.f; }

__global__ __launch_bounds__(256) void k_topk(const float* __restrict__ probs,
                                              float* __restrict__ wdense,
                                              float* __restrict__ usage) {
  int t = blockIdx.x * 256 + threadIdx.x;
  if (t >= TT) return;
  float p[EE];
#pragma unroll
  for (int e = 0; e < EE; ++e) p[e] = probs[t * EE + e];
  int i1 = 0; float v1 = p[0];
#pragma unroll
  for (int e = 1; e < EE; ++e) if (p[e] > v1) { v1 = p[e]; i1 = e; }
  int i2 = -1; float v2 = -1e30f;
#pragma unroll
  for (int e = 0; e < EE; ++e) if (e != i1 && p[e] > v2) { v2 = p[e]; i2 = e; }
  float inv = frcp(v1 + v2);
#pragma unroll
  for (int e = 0; e < EE; ++e)
    wdense[t * EE + e] = (e == i1) ? v1 * inv : ((e == i2) ? v2 * inv : 0.f);
  atomicAdd(&usage[i1], 1.0f);
  atomicAdd(&usage[i2], 1.0f);
}

// ---------------- host launcher ----------------
extern "C" void kernel_launch(void* const* d_in, const int* in_sizes, int n_in,
                              void* d_out, int out_size, void* d_ws, size_t ws_size,
                              hipStream_t stream) {
  (void)in_sizes; (void)n_in; (void)out_size; (void)ws_size;
  const float* x        = (const float*)d_in[0];
  const float* cosb     = (const float*)d_in[1];
  const float* sinb     = (const float*)d_in[2];
  const float* mask     = (const float*)d_in[3];
  const float* ln1_w    = (const float*)d_in[4];
  const float* ln2_w    = (const float*)d_in[5];
  const float* kv_down  = (const float*)d_in[6];
  const float* kv_norm  = (const float*)d_in[7];
  const float* w_uk     = (const float*)d_in[8];
  const float* w_ur     = (const float*)d_in[9];
  const float* w_uv     = (const float*)d_in[10];
  const float* q_down   = (const float*)d_in[11];
  const float* q_norm   = (const float*)d_in[12];
  const float* w_uq     = (const float*)d_in[13];
  const float* w_qr     = (const float*)d_in[14];
  const float* o_proj   = (const float*)d_in[15];
  const float* sh_gate  = (const float*)d_in[16];
  const float* sh_up    = (const float*)d_in[17];
  const float* sh_down  = (const float*)d_in[18];
  const float* r_gate   = (const float*)d_in[19];
  const float* r_up     = (const float*)d_in[20];
  const float* r_down   = (const float*)d_in[21];
  const float* router_w = (const float*)d_in[22];
  const float* router_b = (const float*)d_in[23];

  float* outp  = (float*)d_out;                   // [T,H]
  float* usage = (float*)d_out + (size_t)TT * HH; // [E]

  char* wsb = (char*)d_ws;
  size_t off = 0;
  auto alloc = [&](size_t bytes) -> void* {
    off = (off + 255) & ~(size_t)255;
    void* p = wsb + off;
    off += bytes;
    return p;
  };
  __bf16* xi_bf   = (__bf16*)alloc((size_t)TT * HH * 2);
  float*  ckv_raw = (float*) alloc((size_t)TT * KVR * 4);
  __bf16* ckv_bf  = (__bf16*)alloc((size_t)TT * KVR * 2);
  float*  cq_raw  = (float*) alloc((size_t)TT * QR * 4);
  __bf16* cq_bf   = (__bf16*)alloc((size_t)TT * QR * 2);
  float*  knope   = (float*) alloc((size_t)TT * 512 * 4);
  float*  krope   = (float*) alloc((size_t)TT * 512 * 4);
  float*  vfull   = (float*) alloc((size_t)TT * 1024 * 4);
  float*  qnope   = (float*) alloc((size_t)TT * 512 * 4);
  float*  qrope   = (float*) alloc((size_t)TT * 512 * 4);
  __bf16* qbf     = (__bf16*)alloc((size_t)TT * NHD * HD * 2);
  __bf16* kbf     = (__bf16*)alloc((size_t)TT * NHD * HD * 2);
  __bf16* vbf     = (__bf16*)alloc((size_t)TT * NHD * HD * 2);
  __bf16* obf     = (__bf16*)alloc((size_t)TT * 1024 * 2);
  float*  hnf     = (float*) alloc((size_t)TT * HH * 4);
  __bf16* hnbf    = (__bf16*)alloc((size_t)TT * HH * 2);
  float*  probs   = (float*) alloc((size_t)TT * EE * 4);
  float*  wdense  = (float*) alloc((size_t)TT * EE * 4);
  __bf16* act_bf  = (__bf16*)alloc((size_t)TT * II * 2);
  __bf16* kvd_bf  = (__bf16*)alloc((size_t)HH * KVR * 2);
  __bf16* qd_bf   = (__bf16*)alloc((size_t)HH * QR * 2);
  __bf16* uk_bf   = (__bf16*)alloc((size_t)KVR * 512 * 2);
  __bf16* ur_bf   = (__bf16*)alloc((size_t)KVR * 512 * 2);
  __bf16* uv_bf   = (__bf16*)alloc((size_t)KVR * 1024 * 2);
  __bf16* uq_bf   = (__bf16*)alloc((size_t)QR * 512 * 2);
  __bf16* qr_bf   = (__bf16*)alloc((size_t)QR * 512 * 2);
  __bf16* opj_bf  = (__bf16*)alloc((size_t)HH * HH * 2);
  __bf16* shg_bf  = (__bf16*)alloc((size_t)ESH * HH * II * 2);
  __bf16* shu_bf  = (__bf16*)alloc((size_t)ESH * HH * II * 2);
  __bf16* shd_bf  = (__bf16*)alloc((size_t)ESH * II * HH * 2);
  __bf16* rg_bf   = (__bf16*)alloc((size_t)EE * HH * II * 2);
  __bf16* ru_bf   = (__bf16*)alloc((size_t)EE * HH * II * 2);
  __bf16* rd_bf   = (__bf16*)alloc((size_t)EE * II * HH * 2);

  auto conv = [&](const float* s, __bf16* d, size_t n) {
    unsigned g = (unsigned)((n + 255) / 256);
    if (g > 2048) g = 2048;
    k_f32_to_bf16<<<g, 256, 0, stream>>>(s, d, n);
  };
  auto gemm = [&](int mode, const __bf16* A, int lda, const __bf16* B, int ldb,
                  float* C, int ldc, int M, int N, int K,
                  const float* addsrc, int ldadd, const float* rs, int rstride) {
    dim3 g((unsigned)(M / 128), (unsigned)(N / 64));
    if (mode == 0)
      k_gemm_bf16<0><<<g, 256, 0, stream>>>(A, lda, B, ldb, C, ldc, M, N, K, addsrc, ldadd, rs, rstride);
    else if (mode == 1)
      k_gemm_bf16<1><<<g, 256, 0, stream>>>(A, lda, B, ldb, C, ldc, M, N, K, addsrc, ldadd, rs, rstride);
    else
      k_gemm_bf16<2><<<g, 256, 0, stream>>>(A, lda, B, ldb, C, ldc, M, N, K, addsrc, ldadd, rs, rstride);
  };

  // weight conversions
  conv(kv_down, kvd_bf, (size_t)HH * KVR);
  conv(q_down,  qd_bf,  (size_t)HH * QR);
  conv(w_uk,    uk_bf,  (size_t)KVR * 512);
  conv(w_ur,    ur_bf,  (size_t)KVR * 512);
  conv(w_uv,    uv_bf,  (size_t)KVR * 1024);
  conv(w_uq,    uq_bf,  (size_t)QR * 512);
  conv(w_qr,    qr_bf,  (size_t)QR * 512);
  conv(o_proj,  opj_bf, (size_t)HH * HH);
  conv(sh_gate, shg_bf, (size_t)ESH * HH * II);
  conv(sh_up,   shu_bf, (size_t)ESH * HH * II);
  conv(sh_down, shd_bf, (size_t)ESH * II * HH);
  conv(r_gate,  rg_bf,  (size_t)EE * HH * II);
  conv(r_up,    ru_bf,  (size_t)EE * HH * II);
  conv(r_down,  rd_bf,  (size_t)EE * II * HH);

  // ---- MLA attention ----
  k_rms<<<TT, 256, 0, stream>>>(x, ln1_w, xi_bf, nullptr, HH);
  gemm(0, xi_bf, HH, kvd_bf, KVR, ckv_raw, KVR, TT, KVR, HH, nullptr, 0, nullptr, 0);
  k_rms<<<TT, 256, 0, stream>>>(ckv_raw, kv_norm, ckv_bf, nullptr, KVR);
  gemm(0, xi_bf, HH, qd_bf, QR, cq_raw, QR, TT, QR, HH, nullptr, 0, nullptr, 0);
  k_rms<<<TT, 256, 0, stream>>>(cq_raw, q_norm, cq_bf, nullptr, QR);

  gemm(0, ckv_bf, KVR, uk_bf, 512,  knope, 512,  TT, 512,  KVR, nullptr, 0, nullptr, 0);
  gemm(0, ckv_bf, KVR, ur_bf, 512,  krope, 512,  TT, 512,  KVR, nullptr, 0, nullptr, 0);
  gemm(0, ckv_bf, KVR, uv_bf, 1024, vfull, 1024, TT, 1024, KVR, nullptr, 0, nullptr, 0);
  gemm(0, cq_bf,  QR,  uq_bf, 512,  qnope, 512,  TT, 512,  QR,  nullptr, 0, nullptr, 0);
  gemm(0, cq_bf,  QR,  qr_bf, 512,  qrope, 512,  TT, 512,  QR,  nullptr, 0, nullptr, 0);

  {
    unsigned g = (unsigned)(((size_t)TT * NHD * HD) / 256);
    k_assemble_qk<<<g, 256, 0, stream>>>(qnope, qrope, cosb, sinb, qbf);
    k_assemble_qk<<<g, 256, 0, stream>>>(knope, krope, cosb, sinb, kbf);
    k_assemble_v<<<g, 256, 0, stream>>>(vfull, vbf);
  }

  {
    dim3 g(SS / 128, NHD, BB);
    k_mla_attn<<<g, 256, 0, stream>>>(qbf, kbf, vbf, mask, obf);
  }

  // h = x + o @ o_proj  -> straight into d_out
  gemm(1, obf, 1024, opj_bf, HH, outp, HH, TT, HH, 1024, x, HH, nullptr, 0);

  // ---- MoE ----
  k_rms<<<TT, 256, 0, stream>>>(outp, ln2_w, hnbf, hnf, HH);
  k_router<<<(TT * EE) / 256, 256, 0, stream>>>(hnf, router_w, router_b, probs);
  k_zero8<<<1, 32, 0, stream>>>(usage);
  k_topk<<<TT / 256, 256, 0, stream>>>(probs, wdense, usage);

  for (int ex = 0; ex < ESH + EE; ++ex) {
    const __bf16 *gw, *uw, *dw;
    const float* rs; int rstride;
    if (ex < ESH) {
      gw = shg_bf + (size_t)ex * HH * II;
      uw = shu_bf + (size_t)ex * HH * II;
      dw = shd_bf + (size_t)ex * II * HH;
      rs = nullptr; rstride = 0;
    } else {
      int e = ex - ESH;
      gw = rg_bf + (size_t)e * HH * II;
      uw = ru_bf + (size_t)e * HH * II;
      dw = rd_bf + (size_t)e * II * HH;
      rs = wdense + e; rstride = EE;
    }
    {
      dim3 g(TT / 128, II / 64);
      k_gemm_dual_silu<<<g, 256, 0, stream>>>(hnbf, HH, gw, uw, II, act_bf, II, TT, II, HH);
    }
    gemm(2, act_bf, II, dw, HH, outp, HH, TT, HH, II, nullptr, 0, rs, rstride);
  }
}
